// ImprovedMambaBlock_15994458211472
// MI455X (gfx1250) — matrix operand order repaired and checked by reading, not compiled, not run on hardware
//
#include <hip/hip_runtime.h>
#include <hip/hip_bf16.h>
#include <cmath>

#define D_MODEL 768
#define D_STATE 16
#define D_CONV  4
#define D_INNER 1536
#define DT_RANK 48
#define DT_PAD  64          // dt_lo K padded to multiple of 32
#define BATCH   4
#define SEQLEN  2048
#define NTOK    (BATCH * SEQLEN)   // 8192 tokens

typedef __bf16 bf16_t;
typedef __attribute__((ext_vector_type(16))) __bf16 v16bf;
typedef __attribute__((ext_vector_type(8)))  float  v8f;

union AFrag { v16bf v; bf16_t e[16]; };
union CFrag { v8f   v; float  e[8];  };

// ---------------------------------------------------------------------------
// Weight convert: W [E rows, D cols] fp32 -> Bpack bf16, fragment-native:
//   Bpack[((kt*Npad + n)*32 + kk)] = (n<E && kt*32+kk<D) ? W[n*D + kt*32+kk] : 0
// K-contiguous per column so LDS B fragments load as 2x ds_load_b128.
// ---------------------------------------------------------------------------
__global__ void k_convert_pack(const float* __restrict__ W, bf16_t* __restrict__ out,
                               int E, int D, int Npad, int Kpad) {
    int idx = blockIdx.x * 256 + threadIdx.x;
    int total = (Kpad >> 5) * Npad * 32;
    if (idx >= total) return;
    int kk = idx & 31;
    int n  = (idx >> 5) % Npad;
    int kt = idx / (32 * Npad);
    int k  = kt * 32 + kk;
    float v = (n < E && k < D) ? W[(size_t)n * D + k] : 0.0f;
    out[idx] = (bf16_t)v;
}

// ---------------------------------------------------------------------------
// LayerNorm over D_MODEL=768, one 256-thread block per token, bf16 output
// ---------------------------------------------------------------------------
__global__ __launch_bounds__(256)
void k_layernorm(const float* __restrict__ x, const float* __restrict__ gamma,
                 const float* __restrict__ beta, bf16_t* __restrict__ xn) {
    __shared__ float s1[256], s2[256];
    int t = blockIdx.x;
    const float* row = x + (size_t)t * D_MODEL;
    float a0 = row[threadIdx.x];
    float a1 = row[threadIdx.x + 256];
    float a2 = row[threadIdx.x + 512];
    s1[threadIdx.x] = a0 + a1 + a2;
    s2[threadIdx.x] = a0 * a0 + a1 * a1 + a2 * a2;
    __syncthreads();
    for (int off = 128; off > 0; off >>= 1) {
        if (threadIdx.x < off) {
            s1[threadIdx.x] += s1[threadIdx.x + off];
            s2[threadIdx.x] += s2[threadIdx.x + off];
        }
        __syncthreads();
    }
    float mean = s1[0] * (1.0f / D_MODEL);
    float var  = s2[0] * (1.0f / D_MODEL) - mean * mean;
    float rstd = rsqrtf(var + 1e-5f);
    bf16_t* orow = xn + (size_t)t * D_MODEL;
    int i0 = threadIdx.x, i1 = threadIdx.x + 256, i2 = threadIdx.x + 512;
    orow[i0] = (bf16_t)((a0 - mean) * rstd * gamma[i0] + beta[i0]);
    orow[i1] = (bf16_t)((a1 - mean) * rstd * gamma[i1] + beta[i1]);
    orow[i2] = (bf16_t)((a2 - mean) * rstd * gamma[i2] + beta[i2]);
}

// ---------------------------------------------------------------------------
// GEMM: C[M,N] (fp32, ldc) = A[M,Kpad] (bf16 rm) * Bpack (fragment-native)
// Preconditions: M % 128 == 0, Kpad % 32 == 0, Npad % 128 == 0, A zero-padded
// in K, Bpack zero-padded in N/K.  Epilogue mode: 0=none, 1=+bias, 2=+bias,
// softplus.  Block tile 128x128, 8 waves, wave tile 32x64 (2x4 WMMA frags).
// ---------------------------------------------------------------------------
#define BM 128
#define BN 128
#define BK 32

__global__ __launch_bounds__(256)
void k_gemm_bf16(const bf16_t* __restrict__ A, const bf16_t* __restrict__ Bp,
                 float* __restrict__ C, const float* __restrict__ bias,
                 int M, int N, int Npad, int Kpad, int ldc, int mode) {
    __shared__ bf16_t As[BM][BK + 8];
    __shared__ bf16_t BsT[BN][BK + 8];
    int tid  = threadIdx.x;
    int lane = tid & 31, wid = tid >> 5;
    int wm = wid & 3, wn = wid >> 2;          // 4 waves in M, 2 waves in N
    int blockM = blockIdx.y * BM;
    int blockN = blockIdx.x * BN;
    int hl = lane >> 4, l16 = lane & 15;

    CFrag acc[2][4];
    for (int i = 0; i < 2; i++)
        for (int j = 0; j < 4; j++)
            for (int r = 0; r < 8; r++) acc[i][j].e[r] = 0.0f;

    int nkt = Kpad >> 5;
    for (int kt = 0; kt < nkt; kt++) {
        int k0 = kt * BK;
        // ---- stage A tile (128 x 32): unconditional coalesced uint4 copies
        for (int it = 0; it < 2; it++) {
            int idx = it * 256 + tid;
            int row = idx >> 2;
            int ck  = (idx & 3) * 8;
            const bf16_t* src = A + (size_t)(blockM + row) * Kpad + k0 + ck;
            uint4 tv = *(const uint4*)src;
            if (kt + 1 < nkt) __builtin_prefetch(src + BK, 0, 0);
            *(uint4*)&As[row][ck] = tv;
        }
        // ---- stage B tile (K-major per column): unconditional copies
        for (int it = 0; it < 2; it++) {
            int idx = it * 256 + tid;
            int n   = idx >> 2;
            int ck  = (idx & 3) * 8;
            const bf16_t* src = Bp + ((size_t)kt * Npad + blockN + n) * 32 + ck;
            uint4 tv = *(const uint4*)src;
            if (kt + 1 < nkt) __builtin_prefetch(src + (size_t)Npad * 32, 0, 0);
            *(uint4*)&BsT[n][ck] = tv;
        }
        __syncthreads();

        // A fragments: 16-bit A 16x32 layout (ISA 7.12.2) — two contiguous
        // 8-element runs per lane -> 2x ds_load_b128 each
        AFrag af[2];
        for (int mf = 0; mf < 2; mf++) {
            int r = wm * 32 + mf * 16 + l16;
            for (int j = 0; j < 8; j++) {
                af[mf].e[j]     = As[r][hl * 8 + j];
                af[mf].e[8 + j] = As[r][16 + hl * 8 + j];
            }
        }
        // B fragments: same K layout as A — two 8-element runs per lane -> 2x ds_load_b128
        AFrag bfg[4];
        for (int nf = 0; nf < 4; nf++) {
            int c = wn * 64 + nf * 16 + l16;
            for (int j = 0; j < 8; j++) {
                bfg[nf].e[j]     = BsT[c][hl * 8 + j];
                bfg[nf].e[8 + j] = BsT[c][16 + hl * 8 + j];
            }
        }
        for (int mf = 0; mf < 2; mf++)
            for (int nf = 0; nf < 4; nf++)
                acc[mf][nf].v = __builtin_amdgcn_wmma_f32_16x16x32_bf16(
                    false, af[mf].v, false, bfg[nf].v,
                    (short)0, acc[mf][nf].v, false, false);
        __syncthreads();
    }

    // epilogue: f32 C/D layout — VGPR r: lanes0-15 M=r, lanes16-31 M=r+8
    for (int mf = 0; mf < 2; mf++) {
        for (int nf = 0; nf < 4; nf++) {
            int col = blockN + wn * 64 + nf * 16 + l16;
            if (col >= N) continue;     // only trips when N != Npad (N=80 GEMM)
            float bv = bias ? bias[col] : 0.0f;
            for (int r = 0; r < 8; r++) {
                int row = blockM + wm * 32 + mf * 16 + hl * 8 + r;
                float v = acc[mf][nf].e[r] + bv;
                if (mode == 2) v = (v > 20.0f) ? v : log1pf(__expf(v));
                C[(size_t)row * ldc + col] = v;
            }
        }
    }
}

// ---------------------------------------------------------------------------
// Depthwise causal conv (K=4) + SiLU.  xm lives in xz columns [0,1536), ld=3072
// ---------------------------------------------------------------------------
__global__ void k_conv_silu(const float* __restrict__ xz,
                            const float* __restrict__ conv_w,
                            const float* __restrict__ conv_b,
                            float* __restrict__ uf, bf16_t* __restrict__ ub) {
    int idx = blockIdx.x * 256 + threadIdx.x;
    if (idx >= NTOK * D_INNER) return;
    int c = idx % D_INNER;
    int t = idx / D_INNER;
    int l = t % SEQLEN;
    float acc = conv_b[c];
    for (int k = 0; k < D_CONV; k++) {
        int dl = l + k - (D_CONV - 1);
        if (dl >= 0)
            acc += conv_w[c * D_CONV + k] * xz[(size_t)(t + k - (D_CONV - 1)) * 3072 + c];
    }
    float s = acc / (1.0f + __expf(-acc));   // SiLU
    uf[idx] = s;
    ub[idx] = (bf16_t)s;
}

// ---------------------------------------------------------------------------
// Split x_dbl[8192,80] -> dt_lo bf16 [.,DT_PAD] (cols 48..63 zeroed),
//                         Bs fp32 [.,16], Cs fp32 [.,16]
// 96 virtual slots per token: 0-47 dt_lo, 48-63 Bs, 64-79 Cs, 80-95 dt_lo pad
// ---------------------------------------------------------------------------
__global__ void k_split_xdbl(const float* __restrict__ xdbl,
                             bf16_t* __restrict__ dtlo,
                             float* __restrict__ Bsg, float* __restrict__ Csg) {
    int idx = blockIdx.x * 256 + threadIdx.x;
    if (idx >= NTOK * 96) return;
    int j = idx % 96;
    int t = idx / 96;
    if (j < DT_RANK) {
        dtlo[(size_t)t * DT_PAD + j] = (bf16_t)xdbl[(size_t)t * 80 + j];
    } else if (j < DT_RANK + D_STATE) {
        Bsg[(size_t)t * D_STATE + (j - DT_RANK)] = xdbl[(size_t)t * 80 + j];
    } else if (j < 80) {
        Csg[(size_t)t * D_STATE + (j - DT_RANK - D_STATE)] = xdbl[(size_t)t * 80 + j];
    } else {
        dtlo[(size_t)t * DT_PAD + (j - 32)] = (bf16_t)0.0f;   // zero pad 48..63
    }
}

// ---------------------------------------------------------------------------
// Selective scan. One thread per (b, d) channel; h[16] in registers.
// dt at xz col [0,1536), z at xz col [1536,3072), ld=3072.
// Bs/Cs staged in LDS per 64-step chunk. Fuses +u*D and *silu(z), bf16 out.
// ---------------------------------------------------------------------------
#define TCHUNK 64
__global__ __launch_bounds__(256)
void k_scan(const float* __restrict__ uf, const float* __restrict__ xz,
            const float* __restrict__ Bsg, const float* __restrict__ Csg,
            const float* __restrict__ A_log, const float* __restrict__ Dp,
            bf16_t* __restrict__ yb) {
    __shared__ float sB[TCHUNK][D_STATE];
    __shared__ float sC[TCHUNK][D_STATE];
    int b    = blockIdx.x / (D_INNER / 256);
    int dblk = blockIdx.x % (D_INNER / 256);
    int d    = dblk * 256 + threadIdx.x;

    float Ac[D_STATE], h[D_STATE];
    for (int n = 0; n < D_STATE; n++) {
        Ac[n] = -__expf(A_log[d * D_STATE + n]);
        h[n]  = 0.0f;
    }
    float Dv = Dp[d];
    size_t tokBase = (size_t)b * SEQLEN;

    for (int t0 = 0; t0 < SEQLEN; t0 += TCHUNK) {
        for (int i = threadIdx.x; i < TCHUNK * D_STATE; i += 256) {
            int tt = i >> 4, n = i & 15;
            sB[tt][n] = Bsg[(tokBase + t0 + tt) * D_STATE + n];
            sC[tt][n] = Csg[(tokBase + t0 + tt) * D_STATE + n];
        }
        __syncthreads();
        for (int tt = 0; tt < TCHUNK; tt++) {
            size_t tok = tokBase + t0 + tt;
            float uv  = uf[tok * D_INNER + d];
            float dtv = xz[tok * 3072 + d];
            float zv  = xz[tok * 3072 + 1536 + d];
            float du  = dtv * uv;
            float y = 0.0f;
            for (int n = 0; n < D_STATE; n++) {
                float dA = __expf(dtv * Ac[n]);
                h[n] = dA * h[n] + du * sB[tt][n];
                y += h[n] * sC[tt][n];
            }
            y = (y + uv * Dv) * (zv / (1.0f + __expf(-zv)));
            yb[tok * D_INNER + d] = (bf16_t)y;
        }
        __syncthreads();
    }
}

// ---------------------------------------------------------------------------
// out = residual + mamba_out + sigmoid(gate) * value
// ---------------------------------------------------------------------------
__global__ void k_final(const float* __restrict__ x, const float* __restrict__ mamba,
                        const float* __restrict__ glu, float* __restrict__ out) {
    int idx = blockIdx.x * 256 + threadIdx.x;
    if (idx >= NTOK * D_MODEL) return;
    int d = idx % D_MODEL;
    int t = idx / D_MODEL;
    float g = glu[(size_t)t * 2 * D_MODEL + d];
    float v = glu[(size_t)t * 2 * D_MODEL + D_MODEL + d];
    out[idx] = x[idx] + mamba[idx] + v / (1.0f + __expf(-g));
}

// ---------------------------------------------------------------------------
extern "C" void kernel_launch(void* const* d_in, const int* in_sizes, int n_in,
                              void* d_out, int out_size, void* d_ws, size_t ws_size,
                              hipStream_t stream) {
    (void)in_sizes; (void)n_in; (void)out_size; (void)ws_size;
    const float* x        = (const float*)d_in[0];
    const float* ln_gamma = (const float*)d_in[1];
    const float* ln_beta  = (const float*)d_in[2];
    const float* W_in     = (const float*)d_in[3];
    const float* conv_w   = (const float*)d_in[4];
    const float* conv_b   = (const float*)d_in[5];
    const float* W_x      = (const float*)d_in[6];
    const float* W_dt     = (const float*)d_in[7];
    const float* b_dt     = (const float*)d_in[8];
    const float* A_log    = (const float*)d_in[9];
    const float* Dp       = (const float*)d_in[10];
    const float* W_out    = (const float*)d_in[11];
    const float* W_glu    = (const float*)d_in[12];
    const float* b_glu    = (const float*)d_in[13];

    char* ws = (char*)d_ws;
    size_t off = 0;
    auto alloc = [&](size_t bytes) -> char* {
        char* p = ws + off;
        off += (bytes + 255) & ~(size_t)255;
        return p;
    };
    bf16_t* xn_b   = (bf16_t*)alloc((size_t)NTOK * D_MODEL * 2);
    bf16_t* WinP   = (bf16_t*)alloc((size_t)(768 / 32) * 3072 * 32 * 2);
    bf16_t* WgluP  = (bf16_t*)alloc((size_t)(768 / 32) * 1536 * 32 * 2);
    bf16_t* WxP    = (bf16_t*)alloc((size_t)(1536 / 32) * 128 * 32 * 2);
    bf16_t* WdtP   = (bf16_t*)alloc((size_t)(DT_PAD / 32) * 1536 * 32 * 2);
    bf16_t* WoutP  = (bf16_t*)alloc((size_t)(1536 / 32) * 768 * 32 * 2);
    float*  xz     = (float*) alloc((size_t)NTOK * 3072 * 4);   // xm|z -> dt|z -> glu
    float*  uf     = (float*) alloc((size_t)NTOK * D_INNER * 4); // -> mamba_out
    bf16_t* ub     = (bf16_t*)alloc((size_t)NTOK * D_INNER * 2);
    float*  xdbl   = (float*) alloc((size_t)NTOK * 80 * 4);
    bf16_t* dtlo   = (bf16_t*)alloc((size_t)NTOK * DT_PAD * 2);
    float*  Bsg    = (float*) alloc((size_t)NTOK * D_STATE * 4);
    float*  Csg    = (float*) alloc((size_t)NTOK * D_STATE * 4);
    bf16_t* yb     = (bf16_t*)alloc((size_t)NTOK * D_INNER * 2);
    float*  mamba  = uf;   // alias: uf dead after scan
    float*  glu    = xz;   // alias: xz (dt/z) dead after scan

    dim3 blk(256);
    auto g1 = [](int n) { return dim3((n + 255) / 256); };

    // weight conversion to fragment-native, zero-padded bf16 packs
    k_convert_pack<<<g1(24 * 3072 * 32), blk, 0, stream>>>(W_in,  WinP,  3072, 768,  3072, 768);
    k_convert_pack<<<g1(24 * 1536 * 32), blk, 0, stream>>>(W_glu, WgluP, 1536, 768,  1536, 768);
    k_convert_pack<<<g1(48 * 128 * 32),  blk, 0, stream>>>(W_x,   WxP,   80,   1536, 128,  1536);
    k_convert_pack<<<g1(2 * 1536 * 32),  blk, 0, stream>>>(W_dt,  WdtP,  1536, 48,   1536, DT_PAD);
    k_convert_pack<<<g1(48 * 768 * 32),  blk, 0, stream>>>(W_out, WoutP, 768,  1536, 768,  1536);

    // LayerNorm
    k_layernorm<<<NTOK, blk, 0, stream>>>(x, ln_gamma, ln_beta, xn_b);

    // xz = xn @ W_in^T : [8192, 3072]
    k_gemm_bf16<<<dim3(3072 / BN, NTOK / BM), blk, 0, stream>>>(
        xn_b, WinP, xz, nullptr, NTOK, 3072, 3072, 768, 3072, 0);

    // depthwise conv + SiLU -> u
    k_conv_silu<<<g1(NTOK * D_INNER), blk, 0, stream>>>(xz, conv_w, conv_b, uf, ub);

    // x_dbl = u @ W_x^T : [8192, 80] (Npad=128)
    k_gemm_bf16<<<dim3(1, NTOK / BM), blk, 0, stream>>>(
        ub, WxP, xdbl, nullptr, NTOK, 80, 128, D_INNER, 80, 0);

    // split x_dbl (+ zero-pad dt_lo cols 48..63)
    k_split_xdbl<<<g1(NTOK * 96), blk, 0, stream>>>(xdbl, dtlo, Bsg, Csg);

    // dt = softplus(dt_lo @ W_dt^T + b_dt) -> over xm half of xz (ldc=3072)
    k_gemm_bf16<<<dim3(D_INNER / BN, NTOK / BM), blk, 0, stream>>>(
        dtlo, WdtP, xz, b_dt, NTOK, D_INNER, D_INNER, DT_PAD, 3072, 2);

    // selective scan, fused +u*D and *silu(z) -> y bf16
    k_scan<<<dim3(BATCH * (D_INNER / 256)), blk, 0, stream>>>(
        uf, xz, Bsg, Csg, A_log, Dp, yb);

    // mamba_out = y @ W_out^T : [8192, 768] (into uf region)
    k_gemm_bf16<<<dim3(D_MODEL / BN, NTOK / BM), blk, 0, stream>>>(
        yb, WoutP, mamba, nullptr, NTOK, D_MODEL, D_MODEL, D_INNER, D_MODEL, 0);

    // glu = xn @ W_glu^T + b_glu : [8192, 1536] (into xz region)
    k_gemm_bf16<<<dim3((2 * D_MODEL) / BN, NTOK / BM), blk, 0, stream>>>(
        xn_b, WgluP, glu, b_glu, NTOK, 2 * D_MODEL, 2 * D_MODEL, D_MODEL, 2 * D_MODEL, 1);

    // out = residual + mamba + sigmoid(gate)*value
    k_final<<<g1(NTOK * D_MODEL), blk, 0, stream>>>(x, mamba, glu, (float*)d_out);
}
